// FNO2d_kinet_41171556499508
// MI455X (gfx1250) — matrix operand order repaired and checked by reading, not compiled, not run on hardware
//
#include <hip/hip_runtime.h>
#include <hip/hip_bf16.h>

// ---------------------------------------------------------------------------
// FNO2d-kinet forward for MI455X (gfx1250, wave32, WMMA + TDM).
// Sizes: BS=2, H=W=24, N=576, WIDTH=20, M1=M2=12, NPAIR=165600.
// ---------------------------------------------------------------------------

#define BS     2
#define HH     24
#define NPTS   576            // 24*24
#define CW     20             // width
#define CWP    32             // padded channel rows for p-buffer
#define MM     12             // modes
#define NPAIR  165600         // 576*575/2

typedef __attribute__((ext_vector_type(16))) _Float16     v16h;
typedef __attribute__((ext_vector_type(8)))  float        v8f;
typedef __attribute__((ext_vector_type(4)))  unsigned int u32x4;
typedef __attribute__((ext_vector_type(8)))  int          i32x8;
typedef __attribute__((ext_vector_type(4)))  int          i32x4;

#if defined(__has_builtin)
#if __has_builtin(__builtin_amdgcn_tensor_load_to_lds)
#define HAVE_TDM 1
#endif
#endif
#ifndef HAVE_TDM
#define HAVE_TDM 0
#endif

__device__ __forceinline__ float gelu_exact(float x) {
    return 0.5f * x * (1.0f + erff(x * 0.7071067811865476f));
}

// ---------------------------------------------------------------------------
// TDM: async DMA of a rows x cols f32 tile (row stride in elements) into LDS.
// D# layout per CDNA5 ISA (group0 128b, group1 256b).
// ---------------------------------------------------------------------------
__device__ __forceinline__ void tdm_load_tile_f32(void* lds_dst, const float* gsrc,
                                                  int rows, int cols, int row_stride) {
#if HAVE_TDM
    unsigned lds_off = (unsigned)(unsigned long long)(size_t)lds_dst; // flat low 32 = LDS byte addr
    unsigned long long ga = (unsigned long long)(size_t)gsrc;
    u32x4 g0;
    g0[0] = 1u;                                   // count=1 (valid), user mode
    g0[1] = lds_off;                              // lds_addr (bytes)
    g0[2] = (unsigned)(ga & 0xffffffffull);       // global_addr[31:0]
    g0[3] = (unsigned)((ga >> 32) & 0x1ffffffull) // global_addr[56:32]
          | (2u << 30);                           // type = 2 ("image")
    i32x8 g1;
    g1[0] = (int)(2u << 16);                      // wg_mask=0, data_size=2 (4B)
    g1[1] = (int)(((unsigned)row_stride & 0xffffu) << 16);               // tensor_dim0[15:0]
    g1[2] = (int)((((unsigned)row_stride >> 16) & 0xffffu)               // tensor_dim0[31:16]
          | (((unsigned)rows & 0xffffu) << 16));                         // tensor_dim1[15:0]
    g1[3] = (int)(((unsigned)cols & 0xffffu) << 16);                     // tile_dim0
    g1[4] = rows;                                 // tile_dim1 (tile_dim2=0)
    g1[5] = row_stride;                           // tensor_dim0_stride[31:0]
    g1[6] = 0;                                    // stride hi / dim1_stride lo
    g1[7] = 0;
    i32x4 gz = {0, 0, 0, 0};
#if __clang_major__ >= 23
    i32x8 gz8 = {0, 0, 0, 0, 0, 0, 0, 0};
    __builtin_amdgcn_tensor_load_to_lds(g0, g1, gz, gz, gz8, 0);
#else
    __builtin_amdgcn_tensor_load_to_lds(g0, g1, gz, gz, 0);
#endif
#else
    // Fallback: cooperative vector loads (wave32, one row per lane)
    int lane = threadIdx.x & 31;
    float* dst = (float*)lds_dst;
    if (lane < rows) {
        const float* src = gsrc + lane * row_stride;
        for (int j = 0; j < cols; j += 4) {
            float4 vv = *(const float4*)(src + j);
            dst[lane * cols + j + 0] = vv.x;
            dst[lane * cols + j + 1] = vv.y;
            dst[lane * cols + j + 2] = vv.z;
            dst[lane * cols + j + 3] = vv.w;
        }
    }
#endif
}

__device__ __forceinline__ void tdm_wait() {
#if HAVE_TDM
    __builtin_amdgcn_s_wait_tensorcnt(0);
#endif
}

// --------------------------------------------------------------------------
// 0) Lift: concat(x, gx, gy) @ fc0_w + fc0_b  ->  xbuf[b][c][n]
// --------------------------------------------------------------------------
__global__ void lift_kernel(const float* __restrict__ xin,
                            const float* __restrict__ fc0_w,
                            const float* __restrict__ fc0_b,
                            float* __restrict__ xbuf) {
    int tid = blockIdx.x * blockDim.x + threadIdx.x;
    if (tid >= BS * NPTS) return;
    int n = tid % NPTS, b = tid / NPTS;
    int y = n / HH, xx = n % HH;
    float in[12];
#pragma unroll
    for (int t = 0; t < 10; ++t) in[t] = xin[(b * NPTS + n) * 10 + t];
    in[10] = (float)y  * (1.0f / 23.0f);
    in[11] = (float)xx * (1.0f / 23.0f);
#pragma unroll
    for (int c = 0; c < CW; ++c) {
        float acc = fc0_b[c];
#pragma unroll
        for (int t = 0; t < 12; ++t) acc += in[t] * fc0_w[t * CW + c];
        xbuf[(b * CW + c) * NPTS + n] = acc;
    }
}

// --------------------------------------------------------------------------
// 1a) rfft2 (direct 24-pt DFT; need ky=0..23, kx=0..11)
// --------------------------------------------------------------------------
__global__ void fft_fwd_kernel(const float* __restrict__ xbuf,
                               float* __restrict__ xft) {
    __shared__ float cs[24], sn[24];
    int t = threadIdx.x;
    if (t < 24) {
        float ang = (6.2831853071795864769f / 24.0f) * (float)t;
        __sincosf(ang, &sn[t], &cs[t]);
    }
    __syncthreads();
    int tid = blockIdx.x * blockDim.x + threadIdx.x;
    if (tid >= BS * CW * 24 * MM) return;
    int kx = tid % MM;
    int ky = (tid / MM) % 24;
    int ci = (tid / (MM * 24)) % CW;
    int b  = tid / (MM * 24 * CW);
    const float* X = xbuf + (b * CW + ci) * NPTS;
    float re = 0.0f, im = 0.0f;
    int rowstart = 0;
    for (int y = 0; y < 24; ++y) {
        int idx = rowstart;
        for (int xx = 0; xx < 24; ++xx) {
            float v = X[y * 24 + xx];
            re += v * cs[idx];
            im -= v * sn[idx];
            idx += kx; if (idx >= 24) idx -= 24;
        }
        rowstart += ky; if (rowstart >= 24) rowstart -= 24;
    }
    int o = (((b * CW + ci) * 24 + ky) * MM + kx) * 2;
    xft[o] = re; xft[o + 1] = im;
}

// --------------------------------------------------------------------------
// 1b) mode mix (complex 20x20 per mode; top ky<12 -> w1, bottom -> w2)
// --------------------------------------------------------------------------
__global__ void mix_modes_kernel(const float* __restrict__ xft,
                                 const float* __restrict__ sc_wr,
                                 const float* __restrict__ sc_wi,
                                 float* __restrict__ oft, int k) {
    int tid = blockIdx.x * blockDim.x + threadIdx.x;
    if (tid >= BS * CW * 24 * MM) return;
    int kx = tid % MM;
    int ky = (tid / MM) % 24;
    int o  = (tid / (MM * 24)) % CW;
    int b  = tid / (MM * 24 * CW);
    int sel = (ky >= MM) ? 1 : 0;
    int kyy = ky - sel * MM;
    float or_ = 0.0f, oi_ = 0.0f;
#pragma unroll 4
    for (int i = 0; i < CW; ++i) {
        int xo = (((b * CW + i) * 24 + ky) * MM + kx) * 2;
        float xr = xft[xo], xi = xft[xo + 1];
        long wo = ((((long)(k * 2 + sel) * CW + i) * CW + o) * MM + kyy) * MM + kx;
        float wr = sc_wr[wo], wi = sc_wi[wo];
        or_ += xr * wr - xi * wi;
        oi_ += xr * wi + xi * wr;
    }
    int oo = (((b * CW + o) * 24 + ky) * MM + kx) * 2;
    oft[oo] = or_; oft[oo + 1] = oi_;
}

// --------------------------------------------------------------------------
// 1c) irfft2 (direct): f(0)=1, f(1..11)=2; kx=12 bin is zero
// --------------------------------------------------------------------------
__global__ void fft_inv_kernel(const float* __restrict__ oft,
                               float* __restrict__ abuf) {
    __shared__ float cs[24], sn[24];
    int t = threadIdx.x;
    if (t < 24) {
        float ang = (6.2831853071795864769f / 24.0f) * (float)t;
        __sincosf(ang, &sn[t], &cs[t]);
    }
    __syncthreads();
    int tid = blockIdx.x * blockDim.x + threadIdx.x;
    if (tid >= BS * CW * NPTS) return;
    int xx = tid % HH;
    int y  = (tid / HH) % HH;
    int o  = (tid / NPTS) % CW;
    int b  = tid / (NPTS * CW);
    float acc = 0.0f;
    int rowstart = 0;
    for (int ky = 0; ky < 24; ++ky) {
        int idx = rowstart;
        int base = (((b * CW + o) * 24 + ky) * MM) * 2;
        for (int kx = 0; kx < MM; ++kx) {
            float re = oft[base + kx * 2], im = oft[base + kx * 2 + 1];
            float term = re * cs[idx] - im * sn[idx];
            acc += (kx == 0) ? term : 2.0f * term;
            idx += xx; if (idx >= 24) idx -= 24;
        }
        rowstart += y; if (rowstart >= 24) rowstart -= 24;
    }
    abuf[(b * CW + o) * NPTS + y * HH + xx] = acc * (1.0f / 576.0f);
}

// --------------------------------------------------------------------------
// 2) Channel mix via WMMA + TDM: p[b][o][n] = sum_c W[o][c] x[b][c][n] + bias
//    B tile DMA'd by the tensor data mover; A staged zero-padded in LDS;
//    D written to 32-row padded pbuf -> all fragment/store paths branch-free.
// --------------------------------------------------------------------------
__global__ __launch_bounds__(32) void chanmix_wmma(const float* __restrict__ xbuf,
                                                   const float* __restrict__ w_k,
                                                   const float* __restrict__ w_b,
                                                   float* __restrict__ pbuf, int k) {
    __shared__ float ldsA[16 * 32];   // A tile [M=16][K=32], zero padded
    __shared__ float ldsB[32 * 16];   // B tile [K=32][N=16], zero padded
    __shared__ float ldsBias[CWP];

    int blk = blockIdx.x;                 // 2 batches * 2 mtiles * 36 ntiles
    int b = blk / 72;
    int r = blk % 72;
    int mtile = r / 36;
    int ntile = r % 36;
    int lane = threadIdx.x;
    int half = lane >> 4, l15 = lane & 15;

    // async DMA of B tile (rows = 20 channels, cols = 16 points)
    const float* X = xbuf + b * CW * NPTS;
    tdm_load_tile_f32(ldsB, X + ntile * 16, CW, 16, NPTS);

    // zero pad: B rows 20..31 (192 floats), A tile (512 floats), bias
#pragma unroll
    for (int t = 0; t < 6; ++t) ldsB[CW * 16 + lane + t * 32] = 0.0f;
#pragma unroll
    for (int t = 0; t < 16; ++t) ldsA[lane + t * 32] = 0.0f;
    ldsBias[lane] = (lane < CW) ? w_b[k * CW + lane] : 0.0f;
    __syncthreads();

    // stage A: W[o][c], rows of this mtile (half-wave covers 10 K each)
    const float* W = w_k + k * CW * CW;
    int Mrow = mtile * 16 + l15;
    if (Mrow < CW) {
        const float* src = W + Mrow * CW + half * 10;
        float* dst = ldsA + l15 * 32 + half * 10;
#pragma unroll
        for (int t = 0; t < 10; ++t) dst[t] = src[t];
    }
    tdm_wait();
    __syncthreads();

    v16h a;
#pragma unroll
    for (int e = 0; e < 16; ++e) {
        int v = e >> 1;
        int kk = (v < 4 ? v * 2 : 16 + (v - 4) * 2) + half * 8 + (e & 1);
        a[e] = (_Float16)ldsA[l15 * 32 + kk];
    }
    v16h bf;
#pragma unroll
    for (int e = 0; e < 16; ++e)
        bf[e] = (_Float16)ldsB[((e < 8 ? e : 8 + e) + half * 8) * 16 + l15];

    v8f c = {};
    c = __builtin_amdgcn_wmma_f32_16x16x32_f16(false, a, false, bf,
                                               (short)0, c, false, false);
    int col = ntile * 16 + l15;
    float* P = pbuf + b * CWP * NPTS;
#pragma unroll
    for (int r2 = 0; r2 < 8; ++r2) {
        int o = mtile * 16 + r2 + 8 * half;   // 0..31 (padded rows ok)
        P[o * NPTS + col] = c[r2] + ldsBias[o];
    }
}

// --------------------------------------------------------------------------
// 3) rv update: rv = normalize(rv*ori + ori + s), ori = normalize(rv_init)
// --------------------------------------------------------------------------
__global__ void rv_update_kernel(const float* __restrict__ rv_init,
                                 float* __restrict__ rvbuf,
                                 const float* __restrict__ rv_s, int k) {
    int tid = blockIdx.x * blockDim.x + threadIdx.x;
    if (tid >= BS * NPAIR) return;
    int b  = tid / NPAIR;
    int pr = tid % NPAIR;
    const float s = rv_s[k];
    float ini[CW];
    float ss = 0.0f;
#pragma unroll
    for (int c = 0; c < CW; ++c) {
        float t0 = rv_init[((size_t)(b * CW + c)) * NPAIR + pr];
        ini[c] = t0; ss += t0 * t0;
    }
    float inv = 1.0f / sqrtf(ss);
    float nv[CW];
    float ss2 = 0.0f;
#pragma unroll
    for (int c = 0; c < CW; ++c) {
        float ori = ini[c] * inv;
        float cur = (k == 0) ? ini[c]
                             : rvbuf[((size_t)(b * CW + c)) * NPAIR + pr];
        float t = cur * ori + ori + s;
        nv[c] = t; ss2 += t * t;
    }
    float inv2 = 1.0f / sqrtf(ss2);
#pragma unroll
    for (int c = 0; c < CW; ++c)
        rvbuf[((size_t)(b * CW + c)) * NPAIR + pr] = nv[c] * inv2;
}

// --------------------------------------------------------------------------
// 4a) pairwise norms: vr = |a_i - a_j|, ux = exp(-|p_i - p_j|)
// --------------------------------------------------------------------------
__global__ void pair_r_kernel(const float* __restrict__ pbuf,
                              const float* __restrict__ abuf,
                              float* __restrict__ vr,
                              float* __restrict__ ux) {
    int tid = blockIdx.x * blockDim.x + threadIdx.x;
    if (tid >= BS * NPTS * NPTS) return;
    int b  = tid / (NPTS * NPTS);
    int ij = tid % (NPTS * NPTS);
    int i = ij / NPTS, j = ij % NPTS;
    float sx = 0.0f, sv = 0.0f;
#pragma unroll 4
    for (int c = 0; c < CW; ++c) {
        int op = (b * CWP + c) * NPTS;
        int oa = (b * CW  + c) * NPTS;
        float dx = pbuf[op + i] - pbuf[op + j];
        float dv = abuf[oa + i] - abuf[oa + j];
        sx += dx * dx; sv += dv * dv;
    }
    vr[(size_t)b * NPTS * NPTS + ij] = sqrtf(sv + 1e-12f);
    ux[(size_t)b * NPTS * NPTS + ij] = expf(-sqrtf(sx + 1e-12f));
}

// --------------------------------------------------------------------------
// 4b) v_r_max per batch (v_r >= 0 -> uint-bit atomicMax is order preserving)
// --------------------------------------------------------------------------
__global__ void vrmax_zero_kernel(unsigned* __restrict__ vm) {
    if (threadIdx.x < BS) vm[threadIdx.x] = 0u;
}

__global__ void vrmax_reduce_kernel(const float* __restrict__ vr,
                                    unsigned* __restrict__ vm) {
    __shared__ float red[256];
    int b = blockIdx.y;
    float m = 0.0f;
    for (int idx = blockIdx.x * 256 + threadIdx.x; idx < NPTS * NPTS;
         idx += gridDim.x * 256)
        m = fmaxf(m, vr[(size_t)b * NPTS * NPTS + idx]);
    red[threadIdx.x] = m;
    __syncthreads();
    for (int s = 128; s > 0; s >>= 1) {
        if (threadIdx.x < s)
            red[threadIdx.x] = fmaxf(red[threadIdx.x], red[threadIdx.x + s]);
        __syncthreads();
    }
    if (threadIdx.x == 0) atomicMax(&vm[b], __float_as_uint(red[0]));
}

// --------------------------------------------------------------------------
// 4c) kinet accumulate: x_new[q] = p[q] + v_q + sum_i m_iq*(0.5(v_q - v_i)
//                                 + vr_iq * R[c,i,q]);  GELU for k<3.
// --------------------------------------------------------------------------
__global__ void kinet_kernel(const float* __restrict__ pbuf,
                             const float* __restrict__ abuf,
                             const float* __restrict__ rvbuf,
                             const float* __restrict__ vr,
                             const float* __restrict__ ux,
                             const unsigned* __restrict__ vm,
                             float* __restrict__ xbuf, int k) {
    int tid = blockIdx.x * blockDim.x + threadIdx.x;
    if (tid >= BS * CW * NPTS) return;
    int q = tid % NPTS;
    int c = (tid / NPTS) % CW;
    int b = tid / (NPTS * CW);
    const float invmax = 1.0f / __uint_as_float(vm[b]);
    const float* V  = abuf + (b * CW + c) * NPTS;
    const float* VR = vr + (size_t)b * NPTS * NPTS;
    const float* UX = ux + (size_t)b * NPTS * NPTS;
    const float* RV = rvbuf + (size_t)(b * CW + c) * NPAIR;
    const float vq = V[q];
    float acc = 0.0f;
    for (int i = 0; i < NPTS; ++i) {
        if ((i & 7) == 0 && i + 8 < NPTS)
            __builtin_prefetch(&VR[(size_t)(i + 8) * NPTS + q], 0, 1);
        float vrv = VR[(size_t)i * NPTS + q];
        float m = (vrv * invmax * UX[(size_t)i * NPTS + q] > 0.1f) ? 1.0f : 0.0f;
        float R;
        if (i < q)       R =  RV[(size_t)i * (2 * NPTS - i - 1) / 2 + (q - i - 1)];
        else if (i > q)  R = -RV[(size_t)q * (2 * NPTS - q - 1) / 2 + (i - q - 1)];
        else             R = 0.0f;
        acc += m * (0.5f * (vq - V[i]) + vrv * R);
    }
    float vnew = vq + acc;
    float xn = pbuf[(b * CWP + c) * NPTS + q] + vnew;
    if (k < 3) xn = gelu_exact(xn);
    xbuf[(b * CW + c) * NPTS + q] = xn;
}

// --------------------------------------------------------------------------
// 5) fc1 via WMMA + TDM (M=128, K=20, N=1152) + GELU -> hbuf[o][col]
// --------------------------------------------------------------------------
__global__ __launch_bounds__(32) void fc1_wmma(const float* __restrict__ xbuf,
                                               const float* __restrict__ fc1_w,
                                               const float* __restrict__ fc1_b,
                                               float* __restrict__ hbuf) {
    __shared__ float ldsA[16 * 32];
    __shared__ float ldsB[32 * 16];

    int blk = blockIdx.x;                 // 8 mtiles * 72 ntiles
    int mtile = blk / 72;
    int ntile = blk % 72;
    int lane = threadIdx.x;
    int half = lane >> 4, l15 = lane & 15;

    int bb = ntile / 36;                  // 36 ntiles per batch (576/16)
    int col0 = (ntile % 36) * 16;
    tdm_load_tile_f32(ldsB, xbuf + bb * CW * NPTS + col0, CW, 16, NPTS);

#pragma unroll
    for (int t = 0; t < 6; ++t) ldsB[CW * 16 + lane + t * 32] = 0.0f;
#pragma unroll
    for (int t = 0; t < 16; ++t) ldsA[lane + t * 32] = 0.0f;
    __syncthreads();

    // stage A[o][c] = fc1_w[c][o]; all 128 rows valid, K padded 20->32
    int Mrow = mtile * 16 + l15;
    {
        float* dst = ldsA + l15 * 32 + half * 10;
#pragma unroll
        for (int t = 0; t < 10; ++t)
            dst[t] = fc1_w[(half * 10 + t) * 128 + Mrow];
    }
    tdm_wait();
    __syncthreads();

    v16h a;
#pragma unroll
    for (int e = 0; e < 16; ++e) {
        int v = e >> 1;
        int kk = (v < 4 ? v * 2 : 16 + (v - 4) * 2) + half * 8 + (e & 1);
        a[e] = (_Float16)ldsA[l15 * 32 + kk];
    }
    v16h bf;
#pragma unroll
    for (int e = 0; e < 16; ++e)
        bf[e] = (_Float16)ldsB[((e < 8 ? e : 8 + e) + half * 8) * 16 + l15];

    v8f c = {};
    c = __builtin_amdgcn_wmma_f32_16x16x32_f16(false, a, false, bf,
                                               (short)0, c, false, false);
    int col = bb * NPTS + col0 + l15;     // 0..1151
#pragma unroll
    for (int r2 = 0; r2 < 8; ++r2) {
        int oo = mtile * 16 + r2 + 8 * half;
        float h = gelu_exact(c[r2] + fc1_b[oo]);
        hbuf[oo * (BS * NPTS) + col] = h;
    }
}

// --------------------------------------------------------------------------
// 6) fc2: out[col] = sum_o hbuf[o][col] * fc2_w[o] + fc2_b
// --------------------------------------------------------------------------
__global__ void fc2_kernel(const float* __restrict__ hbuf,
                           const float* __restrict__ fc2_w,
                           const float* __restrict__ fc2_b,
                           float* __restrict__ out) {
    int col = blockIdx.x * blockDim.x + threadIdx.x;
    if (col >= BS * NPTS) return;
    float acc = fc2_b[0];
#pragma unroll 8
    for (int o = 0; o < 128; ++o)
        acc += hbuf[o * (BS * NPTS) + col] * fc2_w[o];
    out[col] = acc;
}

// ---------------------------------------------------------------------------
extern "C" void kernel_launch(void* const* d_in, const int* in_sizes, int n_in,
                              void* d_out, int out_size, void* d_ws, size_t ws_size,
                              hipStream_t stream) {
    (void)in_sizes; (void)n_in; (void)out_size; (void)ws_size;
    const float* xin    = (const float*)d_in[0];
    // d_in[1] = v : dead in the reference graph (never read before overwrite)
    const float* fc0_w  = (const float*)d_in[2];
    const float* fc0_b  = (const float*)d_in[3];
    const float* sc_wr  = (const float*)d_in[4];
    const float* sc_wi  = (const float*)d_in[5];
    const float* w_k    = (const float*)d_in[6];
    const float* w_b    = (const float*)d_in[7];
    const float* fc1_w  = (const float*)d_in[8];
    const float* fc1_b  = (const float*)d_in[9];
    const float* fc2_w  = (const float*)d_in[10];
    const float* fc2_b  = (const float*)d_in[11];
    const float* rv_init= (const float*)d_in[12];
    const float* rv_s   = (const float*)d_in[13];
    float* out = (float*)d_out;

    // workspace layout (floats)
    float* ws    = (float*)d_ws;
    float* xbuf  = ws;                         // 23040
    float* abuf  = xbuf + BS * CW * NPTS;      // 23040
    float* pbuf  = abuf + BS * CW * NPTS;      // 36864 (32-row padded)
    float* xft   = pbuf + BS * CWP * NPTS;     // 23040 (complex interleaved)
    float* oft   = xft  + BS * CW * 24 * MM * 2;
    float* vr    = oft  + BS * CW * 24 * MM * 2;    // 663552
    float* ux    = vr   + (size_t)BS * NPTS * NPTS; // 663552
    float* hbuf  = ux   + (size_t)BS * NPTS * NPTS; // 147456
    unsigned* vm = (unsigned*)(hbuf + 128 * BS * NPTS); // 2
    float* rvbuf = (float*)(vm + 2);                // 6624000

    lift_kernel<<<(BS * NPTS + 255) / 256, 256, 0, stream>>>(xin, fc0_w, fc0_b, xbuf);

    const int nmodes = BS * CW * 24 * MM;           // 11520
    for (int k = 0; k < 4; ++k) {
        fft_fwd_kernel<<<(nmodes + 127) / 128, 128, 0, stream>>>(xbuf, xft);
        mix_modes_kernel<<<(nmodes + 255) / 256, 256, 0, stream>>>(xft, sc_wr, sc_wi, oft, k);
        fft_inv_kernel<<<(BS * CW * NPTS + 255) / 256, 256, 0, stream>>>(oft, abuf);
        chanmix_wmma<<<BS * 2 * 36, 32, 0, stream>>>(xbuf, w_k, w_b, pbuf, k);
        rv_update_kernel<<<(BS * NPAIR + 255) / 256, 256, 0, stream>>>(rv_init, rvbuf, rv_s, k);
        pair_r_kernel<<<(BS * NPTS * NPTS + 255) / 256, 256, 0, stream>>>(pbuf, abuf, vr, ux);
        vrmax_zero_kernel<<<1, 32, 0, stream>>>(vm);
        vrmax_reduce_kernel<<<dim3(32, BS), 256, 0, stream>>>(vr, vm);
        kinet_kernel<<<(BS * CW * NPTS + 255) / 256, 256, 0, stream>>>(
            pbuf, abuf, rvbuf, vr, ux, vm, xbuf, k);
    }

    fc1_wmma<<<8 * 72, 32, 0, stream>>>(xbuf, fc1_w, fc1_b, hbuf);
    fc2_kernel<<<(BS * NPTS + 255) / 256, 256, 0, stream>>>(hbuf, fc2_w, fc2_b, out);
}
